// MambaClassifier_43525198578242
// MI455X (gfx1250) — matrix operand order repaired and checked by reading, not compiled, not run on hardware
//
#include <hip/hip_runtime.h>
#include <hip/hip_bf16.h>
#include <math.h>

typedef __bf16 bf16;
typedef __attribute__((ext_vector_type(16))) __bf16 v16bf;
typedef __attribute__((ext_vector_type(8)))  __bf16 bf16x8;
typedef __attribute__((ext_vector_type(8)))  float  v8f;
typedef __attribute__((ext_vector_type(4)))  unsigned int uint32x4;
typedef __attribute__((ext_vector_type(4)))  int int32x4;
typedef __attribute__((ext_vector_type(8)))  int int32x8;

#define BB      4
#define LL      1024
#define DMODEL  1024
#define DINNER  2048
#define NSTATE  16
#define KCONV   4
#define DTRANK  64
#define NTOK    (BB*LL)   // 4096

static __device__ __forceinline__ float sigmoidf_(float x){ return 1.0f/(1.0f+__expf(-x)); }

// ---- per-lane async copy 16B global -> LDS (ASYNCcnt; ISA 08 §4) ----
static __device__ __forceinline__ void async_b128(unsigned lds_addr, const bf16* gptr){
  asm volatile("global_load_async_to_lds_b128 %0, %1, off"
               :: "v"(lds_addr), "v"(gptr) : "memory");
}
static __device__ __forceinline__ void wait_async0(){
  asm volatile("s_wait_asynccnt 0" ::: "memory");
}
static __device__ __forceinline__ unsigned lds_off(const void* p){
  // Flat shared address: LDS offset is addr[31:0] (ISA §10.2 aperture rules).
  return (unsigned)(size_t)p;
}

// ---- Tensor Data Mover: 2D tile global -> LDS (TENSORcnt; ISA 08 §7-8) ----
// rows x cols(bf16) tile, row stride stride_elems (bf16 elements) in global.
// pad_enable: every 16 DWORDs (one 32-bf16 row) insert 4 DWORDs (8 bf16)
// -> LDS row pitch = 40 bf16 (matches fragment-read layout, conflict-free).
static __device__ __forceinline__ void tdm_load_2d(unsigned lds_addr, const void* gaddr,
                                                   unsigned rows, unsigned cols,
                                                   unsigned stride_elems){
  unsigned long long ga = (unsigned long long)(size_t)gaddr;
  uint32x4 g0;
  g0[0] = 1u;                                              // count=1, user mode
  g0[1] = lds_addr;                                        // LDS byte address
  g0[2] = (unsigned)(ga & 0xFFFFFFFFu);                    // global_addr[31:0]
  g0[3] = (unsigned)((ga >> 32) & 0x01FFFFFFu) | (2u<<30); // addr[56:32] | type=2
  int32x8 g1;
  g1[0] = (int)((1u<<16) | (1u<<20) | (3u<<22) | (3u<<25)); // data_size=2B, pad en, 16DW->+4DW
  g1[1] = (int)((cols & 0xFFFFu) << 16);                    // tensor_dim0[15:0]
  g1[2] = (int)((cols >> 16) | ((rows & 0xFFFFu) << 16));   // dim0[31:16] | dim1[15:0]
  g1[3] = (int)((rows >> 16) | (cols << 16));               // dim1[31:16] | tile_dim0
  g1[4] = (int)(rows & 0xFFFFu);                            // tile_dim1 | tile_dim2=0
  g1[5] = (int)stride_elems;                                // tensor_dim0_stride[31:0]
  g1[6] = 0;                                                // stride0[47:32] | stride1 lo
  g1[7] = 0;
  int32x4 z4 = {0,0,0,0};
#if defined(__clang_major__) && (__clang_major__ >= 23)
  int32x8 z8 = {0,0,0,0,0,0,0,0};
  __builtin_amdgcn_tensor_load_to_lds(g0, g1, z4, z4, z8, 0);
#else
  __builtin_amdgcn_tensor_load_to_lds(g0, g1, z4, z4, 0);
#endif
}

// ---------------- fp32 -> bf16 convert ----------------
__global__ void k_f32_to_bf16(const float* __restrict__ in, bf16* __restrict__ out, int n){
  int i = blockIdx.x*256 + threadIdx.x;
  if (i < n) out[i] = (bf16)in[i];
}

// ---------------- RMSNorm + bf16 (one block per token) ----------------
__global__ void k_rmsnorm_bf16(const float* __restrict__ x, const float* __restrict__ w,
                               bf16* __restrict__ out){
  __shared__ float red[256];
  int tok = blockIdx.x;
  const float* row = x + (size_t)tok * DMODEL;
  float ss = 0.f;
  for (int i = threadIdx.x; i < DMODEL; i += 256){ float v = row[i]; ss += v*v; }
  red[threadIdx.x] = ss; __syncthreads();
  for (int s=128; s>0; s>>=1){ if (threadIdx.x < s) red[threadIdx.x]+=red[threadIdx.x+s]; __syncthreads(); }
  float rms = rsqrtf(red[0]/(float)DMODEL + 1e-5f);
  bf16* orow = out + (size_t)tok*DMODEL;
  for (int i = threadIdx.x; i < DMODEL; i += 256)
    orow[i] = (bf16)(row[i]*rms*w[i]);
}

// =====================================================================
// LDS-staged WMMA GEMM, double buffered:
//   A tile (64x32) : TDM tensor_load_to_lds, hardware-padded to 40-bf16 pitch
//   B tile (128x32): per-lane global_load_async_to_lds_b128
// C[m,n] = sum_k A[m,k]*Bw[n,k]; A:[M,K] bf16, Bw:[N,K] bf16, C:[M,N] f32.
// 8 waves in 2x4 grid, each 32x32 out. MODE: 0 plain, 1 softplus(acc+bias[n]),
// 2 acc+res[m*N+n]. Requires M%64==0, N%128==0, K%32==0.
// =====================================================================
#define MBLK  64
#define NBLK  128
#define KSTEP 32
#define LDA   40   // LDS row pitch (bf16): 20 DWORDs -> conflict-free, 16B aligned

template<int MODE>
__global__ __launch_bounds__(256) void k_wmma_gemm_lds(
    const bf16* __restrict__ A, const bf16* __restrict__ Bw, float* __restrict__ C,
    int M, int N, int Kd, const float* __restrict__ bias, const float* __restrict__ res){
  __shared__ bf16 sA[2][MBLK*LDA];
  __shared__ bf16 sB[2][NBLK*LDA];

  int tid  = threadIdx.x;
  int lane = tid & 31;
  int w    = tid >> 5;
  int wm   = w >> 2;            // 0..1
  int wn   = w & 3;             // 0..3
  int laneL= lane & 15;
  int kh   = lane >> 4;

  int nBlocksN = N / NBLK;
  int mBlk = (blockIdx.x / nBlocksN) * MBLK;
  int nBlk = (blockIdx.x % nBlocksN) * NBLK;

  // B-tile per-thread async assignments (16B chunks, 2 per thread).
  int crow  = tid >> 2;              // 0..63
  int ccol  = (tid & 3) * 8;         // 0,8,16,24 (bf16 elements)
  const bf16* agp  = A  + (size_t)mBlk * Kd;                        // A tile origin (TDM)
  const bf16* bgp0 = Bw + (size_t)(nBlk + crow) * Kd + ccol;        // B rows 0..63
  const bf16* bgp1 = Bw + (size_t)(nBlk + 64 + crow) * Kd + ccol;   // B rows 64..127
  unsigned laT = lds_off(&sA[0][0]);
  unsigned lb0 = lds_off(&sB[0][crow*LDA + ccol]);
  unsigned lb1 = lds_off(&sB[0][(64 + crow)*LDA + ccol]);
  unsigned strA = (unsigned)(MBLK*LDA*sizeof(bf16));
  unsigned strB = (unsigned)(NBLK*LDA*sizeof(bf16));

  // Prologue: fill buffer 0.
  if (w == 0) tdm_load_2d(laT, agp, MBLK, KSTEP, (unsigned)Kd);
  async_b128(lb0, bgp0);
  async_b128(lb1, bgp1);
  wait_async0();
  if (w == 0) __builtin_amdgcn_s_wait_tensorcnt(0);
  __syncthreads();

  v8f acc[2][2] = {};
  int nsteps = Kd / KSTEP;
  for (int s = 0; s < nsteps; ++s){
    int buf = s & 1, nbuf = buf ^ 1;
    if (s + 1 < nsteps){          // issue next tile's fills (overlap with WMMA)
      int ko = (s + 1) * KSTEP;
      if (w == 0) tdm_load_2d(laT + nbuf*strA, agp + ko, MBLK, KSTEP, (unsigned)Kd);
      async_b128(lb0 + nbuf*strB, bgp0 + ko);
      async_b128(lb1 + nbuf*strB, bgp1 + ko);
    }

    // A fragments: 16x32, lane(M=laneL, half kh): K = kh*8..+7 and 16+kh*8..+7.
    v16bf af[2];
    #pragma unroll
    for (int mi = 0; mi < 2; ++mi){
      const bf16* ap = &sA[buf][(wm*32 + mi*16 + laneL)*LDA + kh*8];
      bf16x8 lo = *(const bf16x8*)ap;
      bf16x8 hi = *(const bf16x8*)(ap + 16);
      #pragma unroll
      for (int j=0;j<8;j++){ af[mi][j]=lo[j]; af[mi][j+8]=hi[j]; }
    }
    // B fragments: 32x16, lane(N=laneL, half kh): K = kh*8..+7 and 16+kh*8..+7.
    v16bf bfg[2];
    #pragma unroll
    for (int ni = 0; ni < 2; ++ni){
      const bf16* bp = &sB[buf][(wn*32 + ni*16 + laneL)*LDA + kh*8];
      bf16x8 b0 = *(const bf16x8*)bp;
      bf16x8 b1 = *(const bf16x8*)(bp + 16);
      #pragma unroll
      for (int j=0;j<8;j++){ bfg[ni][j]=b0[j]; bfg[ni][j+8]=b1[j]; }
    }
    #pragma unroll
    for (int mi = 0; mi < 2; ++mi)
      #pragma unroll
      for (int ni = 0; ni < 2; ++ni)
        acc[mi][ni] = __builtin_amdgcn_wmma_f32_16x16x32_bf16(
            false, af[mi], false, bfg[ni], (short)0, acc[mi][ni], false, false);

    wait_async0();                                       // B tile resident
    if (w == 0) __builtin_amdgcn_s_wait_tensorcnt(0);    // A tile resident
    __syncthreads();
  }

  // Epilogue. C/D layout: VGPR r -> M = r + kh*8, N = laneL.
  #pragma unroll
  for (int mi = 0; mi < 2; ++mi){
    #pragma unroll
    for (int ni = 0; ni < 2; ++ni){
      int col = nBlk + wn*32 + ni*16 + laneL;
      #pragma unroll
      for (int r = 0; r < 8; ++r){
        int row = mBlk + wm*32 + mi*16 + r + kh*8;
        float v = acc[mi][ni][r];
        if (MODE==1){ v += bias[col]; v = (v > 20.f) ? v : log1pf(__expf(v)); }
        if (MODE==2){ v += res[(size_t)row*N + col]; }
        C[(size_t)row*N + col] = v;
      }
    }
  }
}

// ---------------- direct WMMA GEMM for small-N (x_proj, N=96) ----------------
__global__ void k_wmma_gemm_small(const bf16* __restrict__ A, const bf16* __restrict__ Bw,
                                  float* __restrict__ C, int M, int N, int Kd){
  int gtid   = blockIdx.x*blockDim.x + threadIdx.x;
  int wave   = gtid >> 5;
  int lane   = threadIdx.x & 31;
  int ntiles = N / 16;
  int mtiles = M / 16;
  if (wave >= mtiles*ntiles) return;   // wave-uniform
  int mt = wave / ntiles, nt = wave % ntiles;
  int laneL = lane & 15, kh = lane >> 4;

  v8f acc = {};
  const bf16* aBase = A  + (size_t)(mt*16 + laneL)*Kd + kh*8;
  const bf16* bBase = Bw + (size_t)(nt*16 + laneL)*Kd + kh*8;
  for (int kk = 0; kk < Kd; kk += 32){
    bf16x8 alo = *(const bf16x8*)(aBase + kk);
    bf16x8 ahi = *(const bf16x8*)(aBase + kk + 16);
    bf16x8 b0  = *(const bf16x8*)(bBase + kk);
    bf16x8 b1  = *(const bf16x8*)(bBase + kk + 16);
    v16bf af, bf_;
    #pragma unroll
    for (int j=0;j<8;j++){ af[j]=alo[j]; af[j+8]=ahi[j]; bf_[j]=b0[j]; bf_[j+8]=b1[j]; }
    acc = __builtin_amdgcn_wmma_f32_16x16x32_bf16(false, af, false, bf_, (short)0, acc, false, false);
  }
  int col = nt*16 + laneL;
  #pragma unroll
  for (int r=0;r<8;r++){
    int row = mt*16 + r + kh*8;
    C[(size_t)row*N + col] = acc[r];
  }
}

// ---------------- depthwise causal conv (K=4) + SiLU ----------------
__global__ void k_conv_silu(const float* __restrict__ xz, const float* __restrict__ cw,
                            const float* __restrict__ cb, float* __restrict__ u,
                            bf16* __restrict__ ub){
  int idx = blockIdx.x*256 + threadIdx.x;          // over NTOK*DINNER
  if (idx >= NTOK*DINNER) return;
  int d   = idx & (DINNER-1);
  int tok = idx >> 11;                              // /2048
  int l   = tok & (LL-1);
  int b   = tok >> 10;
  float s = cb[d];
  #pragma unroll
  for (int t=0;t<KCONV;t++){
    int li = l - (KCONV-1) + t;
    if (li >= 0)
      s += cw[d*KCONV + t] * xz[((size_t)(b*LL + li))*(2*DINNER) + d];
  }
  float v = s * sigmoidf_(s);
  u[idx]  = v;
  ub[idx] = (bf16)v;
}

// ---------------- extract dt slice to bf16 ----------------
__global__ void k_extract_dt(const float* __restrict__ dbc, bf16* __restrict__ dt){
  int i = blockIdx.x*256 + threadIdx.x;             // NTOK*DTRANK
  if (i >= NTOK*DTRANK) return;
  int tok = i >> 6, j = i & 63;
  dt[i] = (bf16)dbc[tok*96 + j];
}

// ---------------- selective scan + gate (one lane per channel) ----------------
__global__ void k_scan(const float* __restrict__ delta, const float* __restrict__ u,
                       const float* __restrict__ dbc, const float* __restrict__ xz,
                       const float* __restrict__ A_log, const float* __restrict__ Dvec,
                       bf16* __restrict__ yg){
  __shared__ float sB[NSTATE], sC[NSTATE];
  int b = blockIdx.x >> 3;                          // 8 blocks of 256 channels / batch
  int d = (blockIdx.x & 7)*256 + threadIdx.x;
  float a[NSTATE], h[NSTATE];
  #pragma unroll
  for (int n=0;n<NSTATE;n++){ a[n] = -__expf(A_log[d*NSTATE+n]); h[n] = 0.f; }
  float Dd = Dvec[d];
  for (int l=0; l<LL; l++){
    int tok = b*LL + l;
    if (threadIdx.x < NSTATE)        sB[threadIdx.x]        = dbc[tok*96 + DTRANK + threadIdx.x];
    else if (threadIdx.x < 2*NSTATE) sC[threadIdx.x-NSTATE] = dbc[tok*96 + DTRANK + NSTATE + (threadIdx.x-NSTATE)];
    __syncthreads();
    float dl = delta[(size_t)tok*DINNER + d];
    float uv = u[(size_t)tok*DINNER + d];
    float du = dl*uv;
    float y  = 0.f;
    #pragma unroll
    for (int n=0;n<NSTATE;n++){
      float dA = __expf(dl * a[n]);
      h[n] = dA*h[n] + du*sB[n];
      y   += h[n]*sC[n];
    }
    y += uv*Dd;
    float z = xz[(size_t)tok*(2*DINNER) + DINNER + d];
    yg[(size_t)tok*DINNER + d] = (bf16)(y * (z * sigmoidf_(z)));
    __syncthreads();
  }
}

// ---------------- final RMSNorm + FC (one block per token) ----------------
__global__ void k_final(const float* __restrict__ outp, const float* __restrict__ normf_w,
                        const float* __restrict__ fc_w, const float* __restrict__ fc_b,
                        float* __restrict__ logits){
  __shared__ float red[256];
  int tok = blockIdx.x;
  const float* row = outp + (size_t)tok*DMODEL;
  float ss = 0.f;
  for (int i=threadIdx.x;i<DMODEL;i+=256){ float v=row[i]; ss += v*v; }
  red[threadIdx.x]=ss; __syncthreads();
  for (int s=128;s>0;s>>=1){ if(threadIdx.x<s) red[threadIdx.x]+=red[threadIdx.x+s]; __syncthreads(); }
  float rms = rsqrtf(red[0]/(float)DMODEL + 1e-5f);
  __syncthreads();
  float acc = 0.f;
  for (int i=threadIdx.x;i<DMODEL;i+=256) acc += row[i]*rms*normf_w[i]*fc_w[i];
  red[threadIdx.x]=acc; __syncthreads();
  for (int s=128;s>0;s>>=1){ if(threadIdx.x<s) red[threadIdx.x]+=red[threadIdx.x+s]; __syncthreads(); }
  if (threadIdx.x==0) logits[tok] = red[0] + fc_b[0];
}

extern "C" void kernel_launch(void* const* d_in, const int* in_sizes, int n_in,
                              void* d_out, int out_size, void* d_ws, size_t ws_size,
                              hipStream_t stream) {
  (void)in_sizes; (void)n_in; (void)out_size; (void)ws_size;
  const float* x         = (const float*)d_in[0];   // [4096,1024]
  const float* in_proj_w = (const float*)d_in[1];   // [4096,1024]
  const float* conv_w    = (const float*)d_in[2];   // [2048,1,4]
  const float* conv_b    = (const float*)d_in[3];   // [2048]
  const float* x_proj_w  = (const float*)d_in[4];   // [96,2048]
  const float* dt_proj_w = (const float*)d_in[5];   // [2048,64]
  const float* dt_proj_b = (const float*)d_in[6];   // [2048]
  const float* A_log     = (const float*)d_in[7];   // [2048,16]
  const float* Dvec      = (const float*)d_in[8];   // [2048]
  const float* out_proj_w= (const float*)d_in[9];   // [1024,2048]
  const float* norm_w    = (const float*)d_in[10];  // [1024]
  const float* normf_w   = (const float*)d_in[11];  // [1024]
  const float* fc_w      = (const float*)d_in[12];  // [1,1024]
  const float* fc_b      = (const float*)d_in[13];  // [1]
  float* logits = (float*)d_out;                    // [4096]

  char* base = (char*)d_ws;
  size_t off = 0;
  auto alloc = [&](size_t bytes)->char*{
    char* p = base + off; off += (bytes + 255) & ~(size_t)255; return p;
  };
  bf16*  wIn   = (bf16*)alloc((size_t)2*DINNER*DMODEL*2);     // [4096,1024] bf16
  bf16*  wXp   = (bf16*)alloc((size_t)96*DINNER*2);           // [96,2048]
  bf16*  wDt   = (bf16*)alloc((size_t)DINNER*DTRANK*2);       // [2048,64]
  bf16*  wOp   = (bf16*)alloc((size_t)DMODEL*DINNER*2);       // [1024,2048]
  bf16*  hbf   = (bf16*)alloc((size_t)NTOK*DMODEL*2);         // normed x, bf16
  float* xz    = (float*)alloc((size_t)NTOK*2*DINNER*4);      // [4096,4096]
  float* u     = (float*)alloc((size_t)NTOK*DINNER*4);
  bf16*  ubf   = (bf16*)alloc((size_t)NTOK*DINNER*2);
  float* dbc   = (float*)alloc((size_t)NTOK*96*4);
  bf16*  dtb   = (bf16*)alloc((size_t)NTOK*DTRANK*2);
  float* delta = (float*)alloc((size_t)NTOK*DINNER*4);
  bf16*  ygb   = (bf16*)alloc((size_t)NTOK*DINNER*2);
  float* outp  = (float*)alloc((size_t)NTOK*DMODEL*4);

  // 1) weights -> bf16
  k_f32_to_bf16<<<(2*DINNER*DMODEL+255)/256, 256, 0, stream>>>(in_proj_w,  wIn, 2*DINNER*DMODEL);
  k_f32_to_bf16<<<(96*DINNER+255)/256,       256, 0, stream>>>(x_proj_w,   wXp, 96*DINNER);
  k_f32_to_bf16<<<(DINNER*DTRANK+255)/256,   256, 0, stream>>>(dt_proj_w,  wDt, DINNER*DTRANK);
  k_f32_to_bf16<<<(DMODEL*DINNER+255)/256,   256, 0, stream>>>(out_proj_w, wOp, DMODEL*DINNER);

  // 2) rmsnorm -> bf16 activations
  k_rmsnorm_bf16<<<NTOK, 256, 0, stream>>>(x, norm_w, hbf);

  // 3) in_proj: xz[4096,4096] = h @ Win^T   (64 x 128 block tiles)
  k_wmma_gemm_lds<0><<<(NTOK/MBLK)*((2*DINNER)/NBLK), 256, 0, stream>>>(
      hbf, wIn, xz, NTOK, 2*DINNER, DMODEL, nullptr, nullptr);

  // 4) causal depthwise conv + SiLU
  k_conv_silu<<<(NTOK*DINNER)/256, 256, 0, stream>>>(xz, conv_w, conv_b, u, ubf);

  // 5) x_proj: dbc[4096,96] = u @ Wx^T  (N=96 -> direct 16-wide tiles; 1536 waves)
  k_wmma_gemm_small<<<(1536*32)/256, 256, 0, stream>>>(ubf, wXp, dbc, NTOK, 96, DINNER);

  // 6) dt slice -> bf16
  k_extract_dt<<<(NTOK*DTRANK)/256, 256, 0, stream>>>(dbc, dtb);

  // 7) dt_proj + softplus: delta[4096,2048]
  k_wmma_gemm_lds<1><<<(NTOK/MBLK)*(DINNER/NBLK), 256, 0, stream>>>(
      dtb, wDt, delta, NTOK, DINNER, DTRANK, dt_proj_b, nullptr);

  // 8) selective scan + silu(z) gate -> yg bf16
  k_scan<<<(BB*DINNER)/256, 256, 0, stream>>>(delta, u, dbc, xz, A_log, Dvec, ygb);

  // 9) out_proj + residual: out[4096,1024]
  k_wmma_gemm_lds<2><<<(NTOK/MBLK)*(DMODEL/NBLK), 256, 0, stream>>>(
      ygb, wOp, outp, NTOK, DMODEL, DINNER, nullptr, x);

  // 10) final rmsnorm + fc -> logits
  k_final<<<NTOK, 256, 0, stream>>>(outp, normf_w, fc_w, fc_b, logits);
}
